// ATDTransformerLayer_47442208752056
// MI455X (gfx1250) — matrix operand order repaired and checked by reading, not compiled, not run on hardware
//
#include <hip/hip_runtime.h>
#include <hip/hip_bf16.h>

// ---------------- problem constants ----------------
#define BB     2
#define HH     192
#define WW2    192
#define NPIX   (HH*WW2)        // 36864 tokens per batch
#define CC     192
#define C3     576
#define NHEADS 6
#define HD     32
#define WS_    16
#define NWTOK  256             // tokens per window
#define NWIN   144             // windows per batch (12x12)
#define NT_    128
#define RD_    20
#define GS_    128
#define NGRP   288             // groups per batch
#define MLP    384
#define TDF    48
#define CH     432
#define ROWS   (BB*NPIX)       // 73728

#define DEV __device__ __forceinline__

typedef __bf16 bf16_t;
typedef __attribute__((ext_vector_type(16))) __bf16 v16bf;
typedef __attribute__((ext_vector_type(8)))  float  v8f;

#define V8ZERO {0.f,0.f,0.f,0.f,0.f,0.f,0.f,0.f}

DEV float bf2f(unsigned short h){ return __uint_as_float(((unsigned int)h) << 16); }
DEV unsigned short f2bf(float f){
  unsigned int u = __float_as_uint(f);
  unsigned int r = u + 0x7FFFu + ((u >> 16) & 1u);
  return (unsigned short)(r >> 16);
}
DEV float gelu_f(float x){ return 0.5f * x * (1.0f + erff(x * 0.7071067811865476f)); }

struct FragAB { union { v16bf v; unsigned int p[8]; }; };

DEV v8f wmma_bf16(const FragAB& a, const FragAB& b, v8f c){
  return __builtin_amdgcn_wmma_f32_16x16x32_bf16(false, a.v, false, b.v, (short)0, c, false, false);
}

// A fragment: 16x32 tile (MxK), row-major source, bounds-guarded
DEV FragAB load_a_g(const unsigned short* __restrict__ A, int lda, int M, int K,
                    int m0, int k0, int lane){
  FragAB f;
  int m = m0 + (lane & 15);
  int half = lane >> 4;
  #pragma unroll
  for (int t = 0; t < 8; ++t){
    int k = k0 + ((t & 3) << 1) + ((t >> 2) << 4) + (half << 3);
    unsigned int u0 = (m < M && k     < K) ? A[(long)m*lda + k]     : 0u;
    unsigned int u1 = (m < M && (k+1) < K) ? A[(long)m*lda + k + 1] : 0u;
    f.p[t] = u0 | (u1 << 16);
  }
  return f;
}
// B fragment: 32x16 tile (KxN), row-major source, bounds-guarded
DEV FragAB load_b_g(const unsigned short* __restrict__ B, int ldb, int K, int N,
                    int k0, int n0, int lane){
  FragAB f;
  int n = n0 + (lane & 15);
  int half = lane >> 4;
  #pragma unroll
  for (int t = 0; t < 8; ++t){
    int k = k0 + ((t & 3) << 1) + ((t >> 2) << 4) + (half << 3);
    unsigned int u0 = (k     < K && n < N) ? B[(long)k*ldb + n]       : 0u;
    unsigned int u1 = ((k+1) < K && n < N) ? B[(long)(k+1)*ldb + n]   : 0u;
    f.p[t] = u0 | (u1 << 16);
  }
  return f;
}

// ---------------- generic batched WMMA GEMM ----------------
// C[M,N] (+)= act(A[M,K] @ B[K,N]); A,B bf16 row-major; C fp32 or bf16.
template<bool ACCUM, int ACT, bool OUTBF>
__global__ __launch_bounds__(128)
void gemm_bf16_kernel(const unsigned short* __restrict__ A, const unsigned short* __restrict__ B,
                      float* __restrict__ Cf, unsigned short* __restrict__ Cb,
                      int M, int N, int K, int lda, int ldb, int ldc,
                      long strideA, long strideB, long strideC)
{
  int lane = threadIdx.x & 31;
  int wave = threadIdx.x >> 5;
  int bz = blockIdx.z;
  const unsigned short* Ab = A + (long)bz * strideA;
  const unsigned short* Bb = B + (long)bz * strideB;
  int m0 = blockIdx.y * 64 + wave * 16;
  int n_base = blockIdx.x * 64;
  if (m0 >= M) return;
  v8f acc[4] = { V8ZERO, V8ZERO, V8ZERO, V8ZERO };
  for (int k0 = 0; k0 < K; k0 += 32){
    FragAB a = load_a_g(Ab, lda, M, K, m0, k0, lane);
    #pragma unroll
    for (int j = 0; j < 4; ++j){
      int n0 = n_base + j * 16;
      if (n0 < N){
        FragAB b = load_b_g(Bb, ldb, K, N, k0, n0, lane);
        acc[j] = wmma_bf16(a, b, acc[j]);
      }
    }
  }
  int nl = lane & 15, half = lane >> 4;
  #pragma unroll
  for (int j = 0; j < 4; ++j){
    int n = n_base + j * 16 + nl;
    if (n >= N) continue;
    #pragma unroll
    for (int r = 0; r < 8; ++r){
      int m = m0 + r + half * 8;
      if (m >= M) continue;
      float v = acc[j][r];
      if (ACT == 1) v = gelu_f(v);
      long off = (long)bz * strideC + (long)m * ldc + n;
      if (OUTBF){ Cb[off] = f2bf(v); }
      else { if (ACCUM) Cf[off] += v; else Cf[off] = v; }
    }
  }
}

// ---------------- elementwise helpers ----------------
__global__ void f2bf_kernel(const float* __restrict__ in, unsigned short* __restrict__ out, long n){
  long i = (long)blockIdx.x * 256 + threadIdx.x;
  if (i < n) out[i] = f2bf(in[i]);
}
__global__ void copy_kernel(const float* __restrict__ in, float* __restrict__ out, long n){
  long i = (long)blockIdx.x * 256 + threadIdx.x;
  if (i < n) out[i] = in[i];
}

// ---------------- LayerNorm (192ch, wave per row) ----------------
__global__ __launch_bounds__(256)
void ln_kernel(const float* __restrict__ x, const float* __restrict__ g, const float* __restrict__ b,
               unsigned short* __restrict__ out, long rows)
{
  int lane = threadIdx.x & 31;
  int wave = threadIdx.x >> 5;
  long row = (long)blockIdx.x * 8 + wave;
  if (row >= rows) return;
  const float* xr = x + row * CC;
  float v[6]; float s = 0.f, s2 = 0.f;
  #pragma unroll
  for (int t = 0; t < 6; ++t){ float f = xr[lane + t * 32]; v[t] = f; s += f; s2 += f * f; }
  #pragma unroll
  for (int off = 16; off; off >>= 1){ s += __shfl_xor(s, off, 32); s2 += __shfl_xor(s2, off, 32); }
  float mean = s * (1.f / CC);
  float var  = s2 * (1.f / CC) - mean * mean;
  float inv  = rsqrtf(var + 1e-5f);
  unsigned short* orow = out + row * CC;
  #pragma unroll
  for (int t = 0; t < 6; ++t){
    int c = lane + t * 32;
    orow[c] = f2bf((v[t] - mean) * inv * g[c] + b[c]);
  }
}

// ---------------- ATD dictionary prep: k-norm, V, td-features ----------------
__global__ __launch_bounds__(64)
void td_prep_kernel(const float* __restrict__ td, const float* __restrict__ wk,
                    const float* __restrict__ wv, const float* __restrict__ fcw,
                    float* __restrict__ kn, unsigned short* __restrict__ vbf,
                    unsigned short* __restrict__ tdfbf)
{
  int bt = blockIdx.x; int b = bt >> 7; int t = bt & 127;
  __shared__ float srow[CC];
  __shared__ float sk[RD_];
  __shared__ float snrm;
  const float* tr = td + ((long)b * NT_ + t) * CC;
  for (int c = threadIdx.x; c < CC; c += 64) srow[c] = tr[c];
  __syncthreads();
  if (threadIdx.x < RD_){
    float a = 0.f;
    for (int c = 0; c < CC; ++c) a += srow[c] * wk[c * RD_ + threadIdx.x];
    sk[threadIdx.x] = a;
  }
  __syncthreads();
  if (threadIdx.x == 0){
    float n2 = 0.f;
    for (int d = 0; d < RD_; ++d) n2 += sk[d] * sk[d];
    snrm = fmaxf(sqrtf(n2), 1e-12f);
  }
  __syncthreads();
  if (threadIdx.x < RD_) kn[((long)b * NT_ + t) * RD_ + threadIdx.x] = sk[threadIdx.x] / snrm;
  for (int j = threadIdx.x; j < CC; j += 64){
    float a = 0.f;
    for (int c = 0; c < CC; ++c) a += srow[c] * wv[c * CC + j];
    vbf[((long)b * NT_ + t) * CC + j] = f2bf(a);
  }
  for (int j = threadIdx.x; j < TDF; j += 64){
    float a = 0.f;
    for (int c = 0; c < CC; ++c) a += srow[c] * fcw[c * TDF + j];
    tdfbf[((long)b * NT_ + t) * TDF + j] = f2bf(a);
  }
}

// ---------------- ATD per-token: q, l2n, softmax sim row, argmax ----------------
__global__ __launch_bounds__(256)
void atd_rows_kernel(const unsigned short* __restrict__ xnbf, const float* __restrict__ wq,
                     const float* __restrict__ kn, const float* __restrict__ scale_p,
                     float* __restrict__ sim, unsigned short* __restrict__ simbf,
                     int* __restrict__ tkid, long rowsTotal)
{
  int lane = threadIdx.x & 31, wave = threadIdx.x >> 5;
  long row = (long)blockIdx.x * 8 + wave;
  if (row >= rowsTotal) return;
  int b = (int)(row / (long)NPIX);
  __shared__ float sq[8][RD_];
  float qd = 0.f;
  if (lane < RD_){
    const unsigned short* xr = xnbf + row * CC;
    for (int c = 0; c < CC; ++c) qd += bf2f(xr[c]) * wq[c * RD_ + lane];
  }
  float n2 = qd * qd;
  #pragma unroll
  for (int off = 16; off; off >>= 1) n2 += __shfl_xor(n2, off, 32);
  float qn = qd / fmaxf(sqrtf(n2), 1e-12f);
  if (lane < RD_) sq[wave][lane] = qn;
  asm volatile("s_wait_dscnt 0" ::: "memory");
  float scp = fminf(fmaxf(scale_p[0], 0.f), 3.f);
  float scl = 1.f + scp * logf(128.f);
  const float* knb = kn + (long)b * NT_ * RD_;
  float svals[4]; float mx = -1e30f;
  #pragma unroll
  for (int t = 0; t < 4; ++t){
    int j = lane + t * 32;
    float a = 0.f;
    #pragma unroll
    for (int d = 0; d < RD_; ++d) a += sq[wave][d] * knb[j * RD_ + d];
    a *= scl; svals[t] = a; mx = fmaxf(mx, a);
  }
  #pragma unroll
  for (int off = 16; off; off >>= 1) mx = fmaxf(mx, __shfl_xor(mx, off, 32));
  float sum = 0.f;
  #pragma unroll
  for (int t = 0; t < 4; ++t){ svals[t] = __expf(svals[t] - mx); sum += svals[t]; }
  #pragma unroll
  for (int off = 16; off; off >>= 1) sum += __shfl_xor(sum, off, 32);
  float inv = 1.f / sum;
  float bestv = -1.f; int besti = 0;
  #pragma unroll
  for (int t = 0; t < 4; ++t){
    int j = lane + t * 32;
    float p = svals[t] * inv;
    sim[row * NT_ + j]   = p;
    simbf[row * NT_ + j] = f2bf(p);
    if (p > bestv){ bestv = p; besti = j; }
  }
  #pragma unroll
  for (int off = 16; off; off >>= 1){
    float ov = __shfl_xor(bestv, off, 32); int oi = __shfl_xor(besti, off, 32);
    if (ov > bestv || (ov == bestv && oi < besti)){ bestv = ov; besti = oi; }
  }
  if (lane == 0) tkid[row] = besti;
}

// ---------------- stable counting argsort of tk_id ----------------
__global__ void hist_kernel(const int* __restrict__ tkid, int* __restrict__ counts, long rowsTotal){
  long i = (long)blockIdx.x * 256 + threadIdx.x;
  if (i < rowsTotal){
    int b = (int)(i / (long)NPIX);
    atomicAdd(&counts[b * NT_ + tkid[i]], 1);
  }
}
__global__ __launch_bounds__(32)
void emit_kernel(const int* __restrict__ tkid, const int* __restrict__ counts, int* __restrict__ idx){
  int blk = blockIdx.x; int b = blk >> 7; int cat = blk & 127;
  int lane = threadIdx.x;
  int offset = 0;
  for (int c = 0; c < cat; ++c) offset += counts[b * NT_ + c];
  const int* ids = tkid + (long)b * NPIX;
  int* ob = idx + (long)b * NPIX;
  for (int i0 = 0; i0 < NPIX; i0 += 32){
    int i = i0 + lane;
    bool m = (ids[i] == cat);
    unsigned long long mask = __ballot(m);
    if (m){
      int pos = offset + (int)__popcll(mask & ((1ull << lane) - 1ull));
      ob[pos] = i;
    }
    offset += (int)__popcll(mask);
  }
}

// ---------------- window attention: (window, head) per workgroup ----------------
__global__ __launch_bounds__(256)
void win_attn_kernel(const unsigned short* __restrict__ qkvbf, const float* __restrict__ rpb,
                     unsigned short* __restrict__ outbf)
{
  __shared__ unsigned short sQ[NWTOK * 32];
  __shared__ unsigned short sK[NWTOK * 32];
  __shared__ unsigned short sV[NWTOK * 32];
  __shared__ unsigned short sP[8][16 * 32];
  int head = blockIdx.x % NHEADS;
  int win  = blockIdx.x / NHEADS;
  int b = win / NWIN;
  int wl = win % NWIN;
  int wy = wl / 12, wx = wl % 12;
  int tid = threadIdx.x;
  int lane = tid & 31, wave = tid >> 5;
  int nl = lane & 15, half = lane >> 4;
  {
    int t = tid;
    int n = (wy * WS_ + (t >> 4)) * WW2 + wx * WS_ + (t & 15);
    const unsigned short* base = qkvbf + ((long)b * NPIX + n) * C3 + head * HD;
    const unsigned int* q4 = (const unsigned int*)(base);
    const unsigned int* k4 = (const unsigned int*)(base + CC);
    const unsigned int* v4 = (const unsigned int*)(base + 2 * CC);
    unsigned int* dq = (unsigned int*)&sQ[t * 32];
    unsigned int* dk = (unsigned int*)&sK[t * 32];
    unsigned int* dv = (unsigned int*)&sV[t * 32];
    #pragma unroll
    for (int u = 0; u < 16; ++u){ dq[u] = q4[u]; dk[u] = k4[u]; dv[u] = v4[u]; }
  }
  __syncthreads();
  const float scale = 0.17677669529663687f; // 1/sqrt(32)
  for (int si = 0; si < 2; ++si){
    int r0 = (wave + si * 8) * 16;
    FragAB qa;
    {
      int m = r0 + nl;
      #pragma unroll
      for (int t2 = 0; t2 < 8; ++t2){
        int k = ((t2 & 3) << 1) + ((t2 >> 2) << 4) + (half << 3);
        qa.p[t2] = *(const unsigned int*)&sQ[m * 32 + k];
      }
    }
    v8f st[16];
    #pragma unroll
    for (int j = 0; j < 16; ++j){
      FragAB kb;
      #pragma unroll
      for (int t2 = 0; t2 < 8; ++t2){
        int kk = ((t2 & 3) << 1) + ((t2 >> 2) << 4) + (half << 3);
        kb.p[t2] = *(const unsigned int*)&sK[(j * 16 + nl) * 32 + kk];
      }
      v8f z = V8ZERO;
      st[j] = wmma_bf16(qa, kb, z);
    }
    // fused scale + relative position bias
    #pragma unroll
    for (int j = 0; j < 16; ++j){
      int q = j * 16 + nl;
      int qy = q >> 4, qx = q & 15;
      #pragma unroll
      for (int r = 0; r < 8; ++r){
        int p = r0 + r + half * 8;
        int py = p >> 4, px = p & 15;
        int rid = (py - qy + 15) * 31 + (px - qx + 15);
        st[j][r] = st[j][r] * scale + rpb[rid * NHEADS + head];
      }
    }
    // row softmax across the 16x256 strip (16-lane half-groups hold a full row)
    #pragma unroll
    for (int r = 0; r < 8; ++r){
      float m = -1e30f;
      #pragma unroll
      for (int j = 0; j < 16; ++j) m = fmaxf(m, st[j][r]);
      #pragma unroll
      for (int off = 8; off; off >>= 1) m = fmaxf(m, __shfl_xor(m, off, 16));
      float s = 0.f;
      #pragma unroll
      for (int j = 0; j < 16; ++j){ float e = __expf(st[j][r] - m); st[j][r] = e; s += e; }
      #pragma unroll
      for (int off = 8; off; off >>= 1) s += __shfl_xor(s, off, 16);
      float inv = 1.f / s;
      #pragma unroll
      for (int j = 0; j < 16; ++j) st[j][r] *= inv;
    }
    // P @ V via wave-private LDS relayout (C-frag -> A-frag)
    v8f o0 = V8ZERO, o1 = V8ZERO;
    unsigned short* myP = sP[wave];
    #pragma unroll
    for (int jj = 0; jj < 8; ++jj){
      #pragma unroll
      for (int t2 = 0; t2 < 2; ++t2){
        #pragma unroll
        for (int r = 0; r < 8; ++r){
          int m = r + half * 8;
          myP[m * 32 + t2 * 16 + nl] = f2bf(st[jj * 2 + t2][r]);
        }
      }
      asm volatile("s_wait_dscnt 0" ::: "memory");
      FragAB pa;
      #pragma unroll
      for (int t2 = 0; t2 < 8; ++t2){
        int k = ((t2 & 3) << 1) + ((t2 >> 2) << 4) + (half << 3);
        pa.p[t2] = *(const unsigned int*)&myP[nl * 32 + k];
      }
      FragAB vb0, vb1;
      #pragma unroll
      for (int t2 = 0; t2 < 8; ++t2){
        int kk = ((t2 & 3) << 1) + ((t2 >> 2) << 4) + (half << 3);
        unsigned int a0 = sV[(jj * 32 + kk) * 32 + nl];
        unsigned int a1 = sV[(jj * 32 + kk + 1) * 32 + nl];
        vb0.p[t2] = a0 | (a1 << 16);
        unsigned int c0 = sV[(jj * 32 + kk) * 32 + 16 + nl];
        unsigned int c1 = sV[(jj * 32 + kk + 1) * 32 + 16 + nl];
        vb1.p[t2] = c0 | (c1 << 16);
      }
      o0 = wmma_bf16(pa, vb0, o0);
      o1 = wmma_bf16(pa, vb1, o1);
    }
    #pragma unroll
    for (int r = 0; r < 8; ++r){
      int p = r0 + r + half * 8;
      int n = (wy * WS_ + (p >> 4)) * WW2 + wx * WS_ + (p & 15);
      unsigned short* ob = outbf + ((long)b * NPIX + n) * CC + head * HD;
      ob[nl]      = f2bf(o0[r]);
      ob[16 + nl] = f2bf(o1[r]);
    }
  }
}

// ---------------- AC-MSA: (batch, group, head) per workgroup, gathered ----------------
__global__ __launch_bounds__(128)
void aca_attn_kernel(const unsigned short* __restrict__ qkvbf, const int* __restrict__ idx,
                     unsigned short* __restrict__ outbf)
{
  __shared__ unsigned short sQ[GS_ * 32];
  __shared__ unsigned short sK[GS_ * 32];
  __shared__ unsigned short sV[GS_ * 32];
  __shared__ unsigned short sP[4][16 * 32];
  __shared__ int sIdx[GS_];
  int head = blockIdx.x % NHEADS;
  int gg   = blockIdx.x / NHEADS;
  int b = gg / NGRP, grp = gg % NGRP;
  int tid = threadIdx.x, lane = tid & 31, wave = tid >> 5;
  int nl = lane & 15, half = lane >> 4;
  {
    int g = idx[(long)b * NPIX + grp * GS_ + tid];
    sIdx[tid] = g;
    const unsigned short* base = qkvbf + ((long)b * NPIX + g) * C3 + head * HD;
    const unsigned int* q4 = (const unsigned int*)(base);
    const unsigned int* k4 = (const unsigned int*)(base + CC);
    const unsigned int* v4 = (const unsigned int*)(base + 2 * CC);
    unsigned int* dq = (unsigned int*)&sQ[tid * 32];
    unsigned int* dk = (unsigned int*)&sK[tid * 32];
    unsigned int* dv = (unsigned int*)&sV[tid * 32];
    #pragma unroll
    for (int u = 0; u < 16; ++u){ dq[u] = q4[u]; dk[u] = k4[u]; dv[u] = v4[u]; }
  }
  __syncthreads();
  const float scale = 0.17677669529663687f;
  for (int si = 0; si < 2; ++si){
    int r0 = (wave + si * 4) * 16;
    FragAB qa;
    {
      int m = r0 + nl;
      #pragma unroll
      for (int t2 = 0; t2 < 8; ++t2){
        int k = ((t2 & 3) << 1) + ((t2 >> 2) << 4) + (half << 3);
        qa.p[t2] = *(const unsigned int*)&sQ[m * 32 + k];
      }
    }
    v8f st[8];
    #pragma unroll
    for (int j = 0; j < 8; ++j){
      FragAB kb;
      #pragma unroll
      for (int t2 = 0; t2 < 8; ++t2){
        int kk = ((t2 & 3) << 1) + ((t2 >> 2) << 4) + (half << 3);
        kb.p[t2] = *(const unsigned int*)&sK[(j * 16 + nl) * 32 + kk];
      }
      v8f z = V8ZERO;
      st[j] = wmma_bf16(qa, kb, z);
    }
    #pragma unroll
    for (int r = 0; r < 8; ++r){
      float m = -1e30f;
      #pragma unroll
      for (int j = 0; j < 8; ++j){ st[j][r] *= scale; m = fmaxf(m, st[j][r]); }
      #pragma unroll
      for (int off = 8; off; off >>= 1) m = fmaxf(m, __shfl_xor(m, off, 16));
      float s = 0.f;
      #pragma unroll
      for (int j = 0; j < 8; ++j){ float e = __expf(st[j][r] - m); st[j][r] = e; s += e; }
      #pragma unroll
      for (int off = 8; off; off >>= 1) s += __shfl_xor(s, off, 16);
      float inv = 1.f / s;
      #pragma unroll
      for (int j = 0; j < 8; ++j) st[j][r] *= inv;
    }
    v8f o0 = V8ZERO, o1 = V8ZERO;
    unsigned short* myP = sP[wave];
    #pragma unroll
    for (int jj = 0; jj < 4; ++jj){
      #pragma unroll
      for (int t2 = 0; t2 < 2; ++t2){
        #pragma unroll
        for (int r = 0; r < 8; ++r){
          int m = r + half * 8;
          myP[m * 32 + t2 * 16 + nl] = f2bf(st[jj * 2 + t2][r]);
        }
      }
      asm volatile("s_wait_dscnt 0" ::: "memory");
      FragAB pa;
      #pragma unroll
      for (int t2 = 0; t2 < 8; ++t2){
        int k = ((t2 & 3) << 1) + ((t2 >> 2) << 4) + (half << 3);
        pa.p[t2] = *(const unsigned int*)&myP[nl * 32 + k];
      }
      FragAB vb0, vb1;
      #pragma unroll
      for (int t2 = 0; t2 < 8; ++t2){
        int kk = ((t2 & 3) << 1) + ((t2 >> 2) << 4) + (half << 3);
        unsigned int a0 = sV[(jj * 32 + kk) * 32 + nl];
        unsigned int a1 = sV[(jj * 32 + kk + 1) * 32 + nl];
        vb0.p[t2] = a0 | (a1 << 16);
        unsigned int c0 = sV[(jj * 32 + kk) * 32 + 16 + nl];
        unsigned int c1 = sV[(jj * 32 + kk + 1) * 32 + 16 + nl];
        vb1.p[t2] = c0 | (c1 << 16);
      }
      o0 = wmma_bf16(pa, vb0, o0);
      o1 = wmma_bf16(pa, vb1, o1);
    }
    #pragma unroll
    for (int r = 0; r < 8; ++r){
      int p = r0 + r + half * 8;
      int g = sIdx[p];
      unsigned short* ob = outbf + ((long)b * NPIX + g) * CC + head * HD;
      ob[nl]      = f2bf(o0[r]);
      ob[16 + nl] = f2bf(o1[r]);
    }
  }
}

// ---------------- depthwise 5x5 conv + GELU + residual (channels-last) ----------------
__global__ __launch_bounds__(256)
void dwconv_kernel(const float* __restrict__ y, const float* __restrict__ dw,
                   unsigned short* __restrict__ zbf)
{
  long i = (long)blockIdx.x * 256 + threadIdx.x;
  long total = (long)ROWS * CH;
  if (i >= total) return;
  int c = (int)(i % CH);
  long bp = i / CH;
  int pix = (int)(bp % NPIX);
  int b = (int)(bp / NPIX);
  int yy = pix / WW2, xx = pix % WW2;
  const float* wc = dw + c * 25;
  float acc = 0.f;
  #pragma unroll
  for (int dy = 0; dy < 5; ++dy){
    int iy = yy + dy - 2;
    if (iy < 0 || iy >= HH) continue;
    #pragma unroll
    for (int dx = 0; dx < 5; ++dx){
      int ix = xx + dx - 2;
      if (ix < 0 || ix >= WW2) continue;
      acc += y[((long)b * NPIX + iy * WW2 + ix) * CH + c] * wc[dy * 5 + dx];
    }
  }
  zbf[i] = f2bf(y[i] + gelu_f(acc));
}

// ---------------- host launch ----------------
extern "C" void kernel_launch(void* const* d_in, const int* in_sizes, int n_in,
                              void* d_out, int out_size, void* d_ws, size_t ws_size,
                              hipStream_t stream)
{
  (void)in_sizes; (void)n_in; (void)out_size; (void)ws_size;
  const float* x      = (const float*)d_in[0];
  const float* td     = (const float*)d_in[1];
  const float* n1g    = (const float*)d_in[2];
  const float* n1b    = (const float*)d_in[3];
  const float* n2g    = (const float*)d_in[4];
  const float* n2b    = (const float*)d_in[5];
  const float* wqkv   = (const float*)d_in[6];
  const float* rpb    = (const float*)d_in[7];
  const float* winp   = (const float*)d_in[8];
  const float* wq     = (const float*)d_in[9];
  const float* wk     = (const float*)d_in[10];
  const float* wv     = (const float*)d_in[11];
  const float* scalep = (const float*)d_in[12];
  const float* acap   = (const float*)d_in[13];
  const float* fcw    = (const float*)d_in[14];
  const float* w1     = (const float*)d_in[15];
  const float* dww    = (const float*)d_in[16];
  const float* w2     = (const float*)d_in[17];
  float* out = (float*)d_out;

  char* p = (char*)d_ws;
  auto alloc = [&](size_t bytes)->char*{
    char* r = p; p += (bytes + 255) & ~(size_t)255; return r;
  };
  unsigned short* xn_bf   = (unsigned short*)alloc((size_t)ROWS * CC * 2);
  unsigned short* qkv_bf  = (unsigned short*)alloc((size_t)ROWS * C3 * 2);
  float*          sim     = (float*)         alloc((size_t)ROWS * NT_ * 4);
  unsigned short* sim_bf  = (unsigned short*)alloc((size_t)ROWS * NT_ * 2);
  unsigned short* tmp_bf  = (unsigned short*)alloc((size_t)ROWS * CC * 2);
  float*          ybuf    = (float*)         alloc((size_t)ROWS * CH * 4);
  unsigned short* z_bf    = (unsigned short*)alloc((size_t)ROWS * CH * 2);
  int*            tk_id   = (int*)           alloc((size_t)ROWS * 4);
  int*            idxb    = (int*)           alloc((size_t)ROWS * 4);
  int*            counts  = (int*)           alloc((size_t)BB * NT_ * 4);
  float*          kn      = (float*)         alloc((size_t)BB * NT_ * RD_ * 4);
  unsigned short* v_bf    = (unsigned short*)alloc((size_t)BB * NT_ * CC * 2);
  unsigned short* tdf_bf  = (unsigned short*)alloc((size_t)BB * NT_ * TDF * 2);
  unsigned short* wqkv_bf = (unsigned short*)alloc((size_t)CC * C3 * 2);
  unsigned short* winp_bf = (unsigned short*)alloc((size_t)CC * CC * 2);
  unsigned short* acap_bf = (unsigned short*)alloc((size_t)CC * CC * 2);
  unsigned short* w1_bf   = (unsigned short*)alloc((size_t)CC * MLP * 2);
  unsigned short* w2_bf   = (unsigned short*)alloc((size_t)CH * CC * 2);

  auto cdiv = [](long a, long b)->long{ return (a + b - 1) / b; };

  // weight conversions
  f2bf_kernel<<<cdiv((long)CC*C3,256),256,0,stream>>>(wqkv, wqkv_bf, (long)CC*C3);
  f2bf_kernel<<<cdiv((long)CC*CC,256),256,0,stream>>>(winp, winp_bf, (long)CC*CC);
  f2bf_kernel<<<cdiv((long)CC*CC,256),256,0,stream>>>(acap, acap_bf, (long)CC*CC);
  f2bf_kernel<<<cdiv((long)CC*MLP,256),256,0,stream>>>(w1, w1_bf, (long)CC*MLP);
  f2bf_kernel<<<cdiv((long)CH*CC,256),256,0,stream>>>(w2, w2_bf, (long)CH*CC);

  // residual accumulator lives in d_out
  copy_kernel<<<cdiv((long)ROWS*CC,256),256,0,stream>>>(x, out, (long)ROWS*CC);

  // LN1 -> xn_bf
  ln_kernel<<<ROWS/8,256,0,stream>>>(x, n1g, n1b, xn_bf, ROWS);

  // qkv = xn @ wqkv  (bf16 out)
  gemm_bf16_kernel<false,0,true><<<dim3(9,1152,1),128,0,stream>>>(
      xn_bf, wqkv_bf, nullptr, qkv_bf, ROWS, C3, CC, CC, C3, C3, 0, 0, 0);

  // ATD prep + per-token sim/argmax
  td_prep_kernel<<<BB*NT_,64,0,stream>>>(td, wk, wv, fcw, kn, v_bf, tdf_bf);
  atd_rows_kernel<<<ROWS/8,256,0,stream>>>(xn_bf, wq, kn, scalep, sim, sim_bf, tk_id, ROWS);

  // x_atd = sim @ V   (batched, accumulate into d_out)
  gemm_bf16_kernel<true,0,false><<<dim3(3,576,BB),128,0,stream>>>(
      sim_bf, v_bf, out, nullptr, NPIX, CC, NT_, NT_, CC, CC,
      (long)NPIX*NT_, (long)NT_*CC, (long)NPIX*CC);

  // x_td = sim @ tdf -> y[:, 384:432]
  gemm_bf16_kernel<false,0,false><<<dim3(1,576,BB),128,0,stream>>>(
      sim_bf, tdf_bf, ybuf + MLP, nullptr, NPIX, TDF, NT_, NT_, TDF, CH,
      (long)NPIX*NT_, (long)NT_*TDF, (long)NPIX*CH);

  // stable counting argsort of tk_id
  hipMemsetAsync(counts, 0, (size_t)BB*NT_*4, stream);
  hist_kernel<<<cdiv(ROWS,256),256,0,stream>>>(tk_id, counts, ROWS);
  emit_kernel<<<BB*NT_,32,0,stream>>>(tk_id, counts, idxb);

  // window attention -> tmp_bf ; proj accumulates
  win_attn_kernel<<<BB*NWIN*NHEADS,256,0,stream>>>(qkv_bf, rpb, tmp_bf);
  gemm_bf16_kernel<true,0,false><<<dim3(3,1152,1),128,0,stream>>>(
      tmp_bf, winp_bf, out, nullptr, ROWS, CC, CC, CC, CC, CC, 0, 0, 0);

  // AC-MSA -> tmp_bf (reuse) ; proj accumulates
  aca_attn_kernel<<<BB*NGRP*NHEADS,128,0,stream>>>(qkv_bf, idxb, tmp_bf);
  gemm_bf16_kernel<true,0,false><<<dim3(3,1152,1),128,0,stream>>>(
      tmp_bf, acap_bf, out, nullptr, ROWS, CC, CC, CC, CC, CC, 0, 0, 0);

  // LN2 -> xn_bf (reuse)
  ln_kernel<<<ROWS/8,256,0,stream>>>(out, n2g, n2b, xn_bf, ROWS);

  // FFN1 with fused GELU -> y[:, 0:384]
  gemm_bf16_kernel<false,1,false><<<dim3(6,1152,1),128,0,stream>>>(
      xn_bf, w1_bf, ybuf, nullptr, ROWS, MLP, CC, CC, MLP, CH, 0, 0, 0);

  // depthwise conv + gelu + residual -> z_bf
  dwconv_kernel<<<cdiv((long)ROWS*CH,256),256,0,stream>>>(ybuf, dww, z_bf);

  // final: z @ w2 accumulates into d_out (K=432, remainder handled by guards)
  gemm_bf16_kernel<true,0,false><<<dim3(3,1152,1),128,0,stream>>>(
      z_bf, w2_bf, out, nullptr, ROWS, CC, CH, CH, CC, CC, 0, 0, 0);
}
